// TripletLossSemiHard_52639119180292
// MI455X (gfx1250) — matrix operand order repaired and checked by reading, not compiled, not run on hardware
//
#include <hip/hip_runtime.h>

typedef __attribute__((ext_vector_type(16))) __bf16 v16bf;
typedef __attribute__((ext_vector_type(8)))  __bf16 v8bf;
typedef __attribute__((ext_vector_type(8)))  float  v8f;
typedef int b128i __attribute__((vector_size(16)));   // matches builtin param type

#define NROWS 8192
#define DIM   512
#define BM    128
#define BN    64
#define KC    32
#define LDST  40   // bf16 elems per LDS row: 80B -> 16B aligned b128, conflict-spread

#define HAS_ASYNC_LDS __has_builtin(__builtin_amdgcn_global_load_async_to_lds_b128)

union Frag { v16bf v; v8bf h[2]; };

// Copy 8 bf16 (16B) from global to LDS: async (no VGPR staging, ASYNCcnt) if available.
__device__ inline void copy16_g2l(const __bf16* __restrict__ g, __bf16* l) {
#if HAS_ASYNC_LDS
  __builtin_amdgcn_global_load_async_to_lds_b128(
      (__attribute__((address_space(1))) b128i*)g,
      (__attribute__((address_space(3))) b128i*)l, 0, 0);
#else
  *(float4*)l = *(const float4*)g;
#endif
}

__device__ inline void wait_async_copies() {
#if HAS_ASYNC_LDS
#if __has_builtin(__builtin_amdgcn_s_wait_asynccnt)
  __builtin_amdgcn_s_wait_asynccnt(0);
#else
  asm volatile("s_wait_asynccnt 0x0" ::: "memory");
#endif
#endif
}

// Order-preserving float -> u32 key, packed with ~rowidx so atomicMax picks
// (largest value, then smallest row index) == argmax first-occurrence.
__device__ inline unsigned long long pack_key(float v, unsigned idx) {
  unsigned u = __float_as_uint(v);
  u = (u & 0x80000000u) ? ~u : (u | 0x80000000u);
  return ((unsigned long long)u << 32) | (unsigned)(~idx);
}

// L2-normalize each row, then split into bf16 hi + bf16 lo (x ~= hi + lo).
__global__ __launch_bounds__(256)
void normalize_split(const float* __restrict__ src,
                     __bf16* __restrict__ dhi, __bf16* __restrict__ dlo,
                     unsigned long long* __restrict__ keys, int init_keys) {
  const int row = blockIdx.x;
  const float* in = src + (size_t)row * DIM;
  __shared__ float red[256];

  float2 v = ((const float2*)in)[threadIdx.x];   // 256 thr * 2 = 512
  red[threadIdx.x] = v.x * v.x + v.y * v.y;
  __syncthreads();
  for (int off = 128; off > 0; off >>= 1) {
    if (threadIdx.x < off) red[threadIdx.x] += red[threadIdx.x + off];
    __syncthreads();
  }
  const float inv = 1.0f / fmaxf(sqrtf(red[0]), 1e-8f);

  float y0 = v.x * inv, y1 = v.y * inv;
  __bf16 h0 = (__bf16)y0, h1 = (__bf16)y1;
  __bf16 l0 = (__bf16)(y0 - (float)h0), l1 = (__bf16)(y1 - (float)h1);
  union { __bf16 b[2]; unsigned u; } ph, pl;
  ph.b[0] = h0; ph.b[1] = h1;
  pl.b[0] = l0; pl.b[1] = l1;
  ((unsigned*)(dhi + (size_t)row * DIM))[threadIdx.x] = ph.u;
  ((unsigned*)(dlo + (size_t)row * DIM))[threadIdx.x] = pl.u;
  if (init_keys && threadIdx.x == 0) keys[row] = 0ULL;
}

// sim[m][n] = bn[m].an[n] via bf16x3 split WMMA; per-column argmax over m,
// diagonal masked; merged globally with packed atomicMax.
__global__ __launch_bounds__(256)
void sim_argmax(const __bf16* __restrict__ anh, const __bf16* __restrict__ anl,
                const __bf16* __restrict__ bnh, const __bf16* __restrict__ bnl,
                unsigned long long* __restrict__ keys) {
  __shared__ __bf16 Ah[2][BM * LDST];   // bn hi tile: M x K (double buffered)
  __shared__ __bf16 Al[2][BM * LDST];   // bn lo tile
  __shared__ __bf16 Bh[2][BN * LDST];   // an hi tile: N x K
  __shared__ __bf16 Bl[2][BN * LDST];   // an lo tile

  const int tid   = threadIdx.x;
  const int lane  = tid & 31;
  const int wave  = tid >> 5;        // 0..7
  const int wm    = wave & 3;        // M sub: 0..3 (32 rows each)
  const int wn    = wave >> 2;       // N sub: 0..1 (32 cols each)
  const int m0    = blockIdx.y * BM;
  const int n0    = blockIdx.x * BN;
  const int lrow  = lane & 15;
  const int hi16  = lane >> 4;       // 0 or 1

  v8f acc[2][2];
  #pragma unroll
  for (int mi = 0; mi < 2; ++mi)
    #pragma unroll
    for (int ni = 0; ni < 2; ++ni) { v8f z = {}; acc[mi][ni] = z; }

  // cooperative-load mapping
  const int arow = tid >> 1;          // 0..127
  const int acol = (tid & 1) * 16;    // 0 / 16 (bf16 elems)
  const int brow = tid >> 2;          // 0..63
  const int bcol = (tid & 3) * 8;     // 0/8/16/24

  const __bf16* gA_h = bnh + (size_t)(m0 + arow) * DIM + acol;
  const __bf16* gA_l = bnl + (size_t)(m0 + arow) * DIM + acol;
  const __bf16* gB_h = anh + (size_t)(n0 + brow) * DIM + bcol;
  const __bf16* gB_l = anl + (size_t)(n0 + brow) * DIM + bcol;

  #define ISSUE_LOADS(buf, kc)                                                 \
    do {                                                                       \
      copy16_g2l(gA_h + (kc),     &Ah[buf][arow * LDST + acol]);               \
      copy16_g2l(gA_h + (kc) + 8, &Ah[buf][arow * LDST + acol + 8]);           \
      copy16_g2l(gA_l + (kc),     &Al[buf][arow * LDST + acol]);               \
      copy16_g2l(gA_l + (kc) + 8, &Al[buf][arow * LDST + acol + 8]);           \
      copy16_g2l(gB_h + (kc),     &Bh[buf][brow * LDST + bcol]);               \
      copy16_g2l(gB_l + (kc),     &Bl[buf][brow * LDST + bcol]);               \
    } while (0)

  ISSUE_LOADS(0, 0);

  int ib = 0;
  for (int kc = 0; kc < DIM; kc += KC, ib ^= 1) {
    wait_async_copies();               // my copies into buf ib are done
    __syncthreads();                   // everyone's copies into buf ib are done
    if (kc + KC < DIM)
      ISSUE_LOADS(ib ^ 1, kc + KC);    // prefetch next chunk into other buffer

    // A-frag per lane: M = base+lrow; K in {kb..kb+7} U {kb+16..kb+23}, kb = hi16*8
    // B-frag per lane: N = base+lrow; K in {kb..kb+7} U {kb+16..kb+23}, kb = hi16*8
    Frag fah[2], fal[2], fbh[2], fbl[2];
    const int kb = hi16 * 8;
    #pragma unroll
    for (int mi = 0; mi < 2; ++mi) {
      const __bf16* pa = &Ah[ib][(wm * 32 + mi * 16 + lrow) * LDST + kb];
      const __bf16* pl = &Al[ib][(wm * 32 + mi * 16 + lrow) * LDST + kb];
      fah[mi].h[0] = *(const v8bf*)pa;       fah[mi].h[1] = *(const v8bf*)(pa + 16);
      fal[mi].h[0] = *(const v8bf*)pl;       fal[mi].h[1] = *(const v8bf*)(pl + 16);
    }
    const int ks = hi16 * 8;
    #pragma unroll
    for (int ni = 0; ni < 2; ++ni) {
      const __bf16* pb = &Bh[ib][(wn * 32 + ni * 16 + lrow) * LDST + ks];
      const __bf16* pl = &Bl[ib][(wn * 32 + ni * 16 + lrow) * LDST + ks];
      fbh[ni].h[0] = *(const v8bf*)pb;       fbh[ni].h[1] = *(const v8bf*)(pb + 16);
      fbl[ni].h[0] = *(const v8bf*)pl;       fbl[ni].h[1] = *(const v8bf*)(pl + 16);
    }

    #pragma unroll
    for (int mi = 0; mi < 2; ++mi)
      #pragma unroll
      for (int ni = 0; ni < 2; ++ni) {
        acc[mi][ni] = __builtin_amdgcn_wmma_f32_16x16x32_bf16(
            false, fah[mi].v, false, fbh[ni].v, (short)0, acc[mi][ni], false, false);
        acc[mi][ni] = __builtin_amdgcn_wmma_f32_16x16x32_bf16(
            false, fah[mi].v, false, fbl[ni].v, (short)0, acc[mi][ni], false, false);
        acc[mi][ni] = __builtin_amdgcn_wmma_f32_16x16x32_bf16(
            false, fal[mi].v, false, fbh[ni].v, (short)0, acc[mi][ni], false, false);
      }
  }
  #undef ISSUE_LOADS

  // Epilogue: per-column argmax over this block's 128 M rows.
  // C-layout: lane holds (M = base + hi16*8 + r, N = base + (lane&15)).
  #pragma unroll
  for (int ni = 0; ni < 2; ++ni) {
    const int col = n0 + wn * 32 + ni * 16 + lrow;
    float best = -3.0f;                // below min possible (sim - 1 >= -2)
    unsigned bestM = 0xFFFFFFFFu;
    #pragma unroll
    for (int mi = 0; mi < 2; ++mi) {
      #pragma unroll
      for (int r = 0; r < 8; ++r) {
        const int mrow = m0 + wm * 32 + mi * 16 + hi16 * 8 + r;
        const float v = acc[mi][ni][r] - ((mrow == col) ? 1.0f : 0.0f);
        if (v > best || (v == best && (unsigned)mrow < bestM)) {
          best = v; bestM = (unsigned)mrow;
        }
      }
    }
    // lanes L and L+16 hold the same column with disjoint M sets
    const float    oBest = __shfl_xor(best, 16, 32);
    const unsigned oM    = (unsigned)__shfl_xor((int)bestM, 16, 32);
    if (oBest > best || (oBest == best && oM < bestM)) { best = oBest; bestM = oM; }
    if (lane < 16)
      atomicMax(keys + col, pack_key(best, bestM));
  }
}

__global__ __launch_bounds__(128)
void gather_rows(const float* __restrict__ b,
                 const unsigned long long* __restrict__ keys,
                 float* __restrict__ out) {
  const int row = blockIdx.x;
  const unsigned idx = ~(unsigned)(keys[row] & 0xFFFFFFFFull);
  const float4* src = (const float4*)(b + (size_t)idx * DIM);
  float4*       dst = (float4*)(out + (size_t)row * DIM);
  dst[threadIdx.x] = src[threadIdx.x];
}

extern "C" void kernel_launch(void* const* d_in, const int* in_sizes, int n_in,
                              void* d_out, int out_size, void* d_ws, size_t ws_size,
                              hipStream_t stream) {
  const float* a = (const float*)d_in[0];
  const float* b = (const float*)d_in[1];

  const size_t NE = (size_t)NROWS * DIM;
  __bf16* anh = (__bf16*)d_ws;
  __bf16* anl = anh + NE;
  __bf16* bnh = anl + NE;
  __bf16* bnl = bnh + NE;
  unsigned long long* keys = (unsigned long long*)(bnl + NE);
  float* out = (float*)d_out;

  normalize_split<<<NROWS, 256, 0, stream>>>(a, anh, anl, keys, 1);
  normalize_split<<<NROWS, 256, 0, stream>>>(b, bnh, bnl, keys, 0);

  dim3 grid(NROWS / BN, NROWS / BM);   // 128 x 64 tile blocks
  sim_argmax<<<grid, 256, 0, stream>>>(anh, anl, bnh, bnl, keys);

  gather_rows<<<NROWS, DIM / 4, 0, stream>>>(b, keys, out);
}
